// TemporalXORReservoirNetwork_66795331387935
// MI455X (gfx1250) — matrix operand order repaired and checked by reading, not compiled, not run on hardware
//
#include <hip/hip_runtime.h>

// ---------------------------------------------------------------------------
// TemporalXORReservoirNetwork for MI455X (gfx1250, wave32, WMMA).
//
// Per step:  V = 0.9*V + x_t@Win.T + S@W ;  S = (V>=1);  V *= (1-S)
// S is binary -> bf16 A operand is exact. W converted once to bf16,
// transposed (Wt[n][k] = W[k][n]) so both WMMA operands load contiguously.
// Wt (8 MB) stays L2-resident across all 200 sequential steps.
//
// Tiling: block = 8 waves (256 thr) arranged 4(M) x 2(N); per-wave tile
// 64x32 (4 A-frags x 2 B-frags -> 8 WMMA / 12 b128 loads). Block tile
// 256x64 covers the full batch -> Wt is streamed exactly once per step
// (8 MB), S re-read by the 32 N-blocks (32 MB) => ~40 MB L2/step.
//
// ws layout (bytes):
//   [0,                8 MB)  Wt   bf16 [2048][2048]   (transposed W)
//   [8 MB,            10 MB)  V    f32  [256][2048]
//   [10 MB,           11 MB)  S0   bf16 [256][2048]    (spike ping)
//   [11 MB,           12 MB)  S1   bf16 [256][2048]    (spike pong)
// ---------------------------------------------------------------------------

#define RES    2048
#define SEQ    200
#define BATCH  256
#define NIN    3

typedef __bf16 bf16_t;
typedef __attribute__((ext_vector_type(16))) __bf16 v16bf;
typedef __attribute__((ext_vector_type(8)))  __bf16 v8bf;
typedef __attribute__((ext_vector_type(8)))  float  v8f;

union V16U { v16bf v; v8bf h[2]; };

// --- one-time: W f32 row-major [k][n]  ->  Wt bf16 row-major [n][k] --------
__global__ __launch_bounds__(256) void wtrans_kernel(const float* __restrict__ W,
                                                     bf16_t* __restrict__ Wt) {
    __shared__ bf16_t tile[32][33];   // +1 pad: no LDS bank conflicts
    const int tx = threadIdx.x;       // 0..31
    const int ty = threadIdx.y;       // 0..7
    const int n0 = blockIdx.x * 32;
    const int k0 = blockIdx.y * 32;
#pragma unroll
    for (int i = 0; i < 32; i += 8)
        tile[ty + i][tx] = (bf16_t)W[(size_t)(k0 + ty + i) * RES + (n0 + tx)];
    __syncthreads();
#pragma unroll
    for (int i = 0; i < 32; i += 8)
        Wt[(size_t)(n0 + ty + i) * RES + (k0 + tx)] = tile[tx][ty + i];
}

// --- one-time: zero V + S0 + S1 (contiguous 4 MB) --------------------------
__global__ void zero4_kernel(float4* __restrict__ p, int n4) {
    int i = blockIdx.x * blockDim.x + threadIdx.x;
    if (i < n4) p[i] = make_float4(0.f, 0.f, 0.f, 0.f);
}

// --- per-timestep fused GEMM + LIF update ----------------------------------
// grid = (RES/64), block = 256 (8 waves: 4 M-groups x 2 N-groups)
// wave computes a 64(M) x 32(N) strip: 4 M-subtiles x 2 N-subtiles.
__global__ __launch_bounds__(256) void step_kernel(
    const bf16_t* __restrict__ Sprev,   // [BATCH][RES] bf16 spikes (t-1)
    bf16_t*       __restrict__ Snext,   // [BATCH][RES] bf16 spikes (t)
    float*        __restrict__ V,       // [BATCH][RES] membrane
    const bf16_t* __restrict__ Wt,      // [RES][RES]  W transposed, bf16
    const float*  __restrict__ x,       // [SEQ][BATCH][NIN]
    const float*  __restrict__ Win,     // [RES][NIN]
    float*        __restrict__ spk_out, // d_out slice for step t
    int t)
{
    const int lane = threadIdx.x & 31;
    const int wave = threadIdx.x >> 5;       // 0..7
    const int hl   = lane >> 4;              // lane half (0/1)
    const int row  = lane & 15;
    const int m0 = (wave >> 1) * 64;                      // 0,64,128,192
    const int n0 = blockIdx.x * 64 + (wave & 1) * 32;

    v8f acc[4][2] = {};

    // A in contiguous k order (not the ISA table order): lanes 0-15 row M=row, K 0-15; lanes 16-31 K 16-31
    const bf16_t* aBase = Sprev + (size_t)(m0 + row) * RES + hl * 16;
    // B in the same contiguous k order as A: lanes 0-15 col N=row, K 0-15; lanes 16-31 K 16-31
    const bf16_t* bBase = Wt + (size_t)(n0 + row) * RES + hl * 16;

    for (int k = 0; k < RES; k += 32) {
        V16U B0, B1;
        B0.h[0] = *(const v8bf*)(bBase + k);
        B0.h[1] = *(const v8bf*)(bBase + k + 8);
        B1.h[0] = *(const v8bf*)(bBase + (size_t)16 * RES + k);
        B1.h[1] = *(const v8bf*)(bBase + (size_t)16 * RES + k + 8);
        __builtin_prefetch(bBase + k + 32, 0, 3);
#pragma unroll
        for (int sm = 0; sm < 4; ++sm) {
            const bf16_t* ap = aBase + (size_t)sm * 16 * RES + k;
            V16U A;
            A.h[0] = *(const v8bf*)(ap);
            A.h[1] = *(const v8bf*)(ap + 8);
            acc[sm][0] = __builtin_amdgcn_wmma_f32_16x16x32_bf16(
                false, A.v, false, B0.v, (short)0, acc[sm][0], false, false);
            acc[sm][1] = __builtin_amdgcn_wmma_f32_16x16x32_bf16(
                false, A.v, false, B1.v, (short)0, acc[sm][1], false, false);
        }
    }

    // fused epilogue: input current + leaky integrate + fire + zero-reset
    const float* xt = x + (size_t)t * BATCH * NIN;
#pragma unroll
    for (int sn = 0; sn < 2; ++sn) {
        const int n = n0 + sn * 16 + row;
        const float w0 = Win[n * 3 + 0];
        const float w1 = Win[n * 3 + 1];
        const float w2 = Win[n * 3 + 2];
#pragma unroll
        for (int sm = 0; sm < 4; ++sm) {
#pragma unroll
            for (int r = 0; r < 8; ++r) {
                const int m = m0 + sm * 16 + hl * 8 + r;   // C layout: VGPR r -> M = r + 8*half
                const float inp = xt[m * 3 + 0] * w0 + xt[m * 3 + 1] * w1 + xt[m * 3 + 2] * w2;
                const size_t idx = (size_t)m * RES + n;
                const float v  = 0.9f * V[idx] + acc[sm][sn][r] + inp;
                const float sp = (v >= 1.0f) ? 1.0f : 0.0f;
                V[idx]       = v * (1.0f - sp);
                spk_out[idx] = sp;
                Snext[idx]   = (bf16_t)sp;
            }
        }
    }
}

// --- readout: logits[b][o] = mean(spk[190:200])[b] @ Wout[o] ---------------
__global__ __launch_bounds__(256) void logits_kernel(const float* __restrict__ spk,
                                                     const float* __restrict__ Wout,
                                                     float* __restrict__ logits) {
    __shared__ float s0[256], s1[256];
    const int b = blockIdx.x, tid = threadIdx.x;
    float a0 = 0.f, a1 = 0.f;
    for (int n = tid; n < RES; n += 256) {
        float avg = 0.f;
#pragma unroll
        for (int t = SEQ - 10; t < SEQ; ++t)
            avg += spk[((size_t)t * BATCH + b) * RES + n];
        avg *= 0.1f;
        a0 += avg * Wout[n];
        a1 += avg * Wout[RES + n];
    }
    s0[tid] = a0; s1[tid] = a1;
    __syncthreads();
    for (int off = 128; off > 0; off >>= 1) {
        if (tid < off) { s0[tid] += s0[tid + off]; s1[tid] += s1[tid + off]; }
        __syncthreads();
    }
    if (tid == 0) { logits[b * 2 + 0] = s0[0]; logits[b * 2 + 1] = s1[0]; }
}

extern "C" void kernel_launch(void* const* d_in, const int* in_sizes, int n_in,
                              void* d_out, int out_size, void* d_ws, size_t ws_size,
                              hipStream_t stream) {
    (void)in_sizes; (void)n_in; (void)out_size; (void)ws_size;
    const float* x    = (const float*)d_in[0];   // [200,256,3]
    const float* W    = (const float*)d_in[1];   // [2048,2048]
    const float* Win  = (const float*)d_in[2];   // [2048,3]
    const float* Wout = (const float*)d_in[3];   // [2,2048]
    float* out = (float*)d_out;                  // spk_rec then logits

    char* ws = (char*)d_ws;
    bf16_t* Wt = (bf16_t*)ws;                                            // 8 MB
    float*  V  = (float*)(ws + (size_t)RES * RES * 2);                   // 2 MB
    bf16_t* S0 = (bf16_t*)(ws + (size_t)RES * RES * 2 + (size_t)BATCH * RES * 4);
    bf16_t* S1 = S0 + (size_t)BATCH * RES;

    // one-time per call: downconvert+transpose W, zero V/S0/S1
    wtrans_kernel<<<dim3(RES / 32, RES / 32), dim3(32, 8), 0, stream>>>(W, Wt);
    const int n4 = BATCH * RES * (4 + 2 + 2) / 16;   // V + S0 + S1 contiguous
    zero4_kernel<<<(n4 + 255) / 256, 256, 0, stream>>>((float4*)V, n4);

    // 200 sequential steps (inter-step sync = kernel boundary)
    for (int t = 0; t < SEQ; ++t) {
        const bf16_t* Sp = (t & 1) ? S1 : S0;
        bf16_t*       Sn = (t & 1) ? S0 : S1;
        step_kernel<<<dim3(RES / 64), 256, 0, stream>>>(
            Sp, Sn, V, Wt, x, Win, out + (size_t)t * BATCH * RES, t);
    }

    logits_kernel<<<BATCH, 256, 0, stream>>>(out, Wout, out + (size_t)SEQ * BATCH * RES);
}
